// EuclideanCodebook_31301721653526
// MI455X (gfx1250) — matrix operand order repaired and checked by reading, not compiled, not run on hardware
//
#include <hip/hip_runtime.h>
#include <hip/hip_bf16.h>

typedef _Float16 v16h __attribute__((ext_vector_type(16)));
typedef _Float16 v8h  __attribute__((ext_vector_type(8)));
typedef float    v8f  __attribute__((ext_vector_type(8)));

#define N_ROWS 65536
#define D_DIM  32
#define K_CODES 4096
#define DECAY  0.1f
#define EPSV   1e-5f
#define NORM_EPS 1e-8f

// ---------------- kernel: zero scratch (bins + embed_sum) ----------------
__global__ void vq_zero(float* __restrict__ p, int n) {
    int i = blockIdx.x * blockDim.x + threadIdx.x;
    if (i < n) p[i] = 0.0f;
}

// ---------------- kernel: wave-per-row L2 normalize ----------------
// one wave32 per row; writes f32 + f16 copies. For the codebook it also writes
// e2m4[k] = 4 - sum(en^2) so the assign kernel's score (s+4) is a single FMA.
__global__ __launch_bounds__(256) void vq_normalize(
    const float* __restrict__ src, float* __restrict__ dstF,
    _Float16* __restrict__ dstH, float* __restrict__ e2m4, int rows)
{
    int wave = (blockIdx.x * blockDim.x + threadIdx.x) >> 5;
    int lane = threadIdx.x & 31;
    if (wave >= rows) return;
    float v = src[wave * D_DIM + lane];
    float ss = v * v;
    #pragma unroll
    for (int m = 16; m >= 1; m >>= 1) ss += __shfl_xor(ss, m);
    float inv = 1.0f / (sqrtf(ss) + NORM_EPS);
    float nv = v * inv;
    dstF[wave * D_DIM + lane] = nv;
    dstH[wave * D_DIM + lane] = (_Float16)nv;
    if (e2m4 != nullptr && lane == 0) e2m4[wave] = 4.0f - ss * inv * inv;
}

// ---------------- kernel: WMMA distance + packed-key argmax ----------------
// block = 256 threads = 8 waves; each wave owns 16 rows; block owns 128 rows.
// One v_wmma_f32_16x16x32_f16 per 16-code tile (contraction D=32 == WMMA K).
// Argmax via monotonic float->uint key packing: 3 VALU per score element
// (v_pk_fma + v_and_or + v_max_u32), software-pipelined past the WMMA hazard.
__global__ __launch_bounds__(256) void vq_assign(
    const _Float16* __restrict__ fnH, const float* __restrict__ fnF,
    const _Float16* __restrict__ enH, const float* __restrict__ enF,
    const float* __restrict__ e2m4,
    float* __restrict__ qOut, float* __restrict__ indOut,
    float* __restrict__ bins, float* __restrict__ embedSum)
{
    __shared__ int codeBuf[128];

    const int tid     = threadIdx.x;
    const int waveId  = tid >> 5;
    const int lane    = tid & 31;
    const int halfSel = lane >> 4;   // 0: lanes 0-15, 1: lanes 16-31
    const int l15     = lane & 15;
    const int rowBase = blockIdx.x * 128 + waveId * 16;

    // ---- A fragment (16x32 f16), contiguous k order (k = 16h + s, same as B):
    // lanes 0-15 : M=lane,    K = 0..15
    // lanes 16-31: M=lane-16, K = 16..31
    const int arow = rowBase + l15;
    const int aoff = halfSel * 16;
    v8h alo = *(const v8h*)(fnH + arow * D_DIM + aoff);
    v8h ahi = *(const v8h*)(fnH + arow * D_DIM + aoff + 8);
    v16h A;
    #pragma unroll
    for (int j = 0; j < 8; ++j) { A[j] = alo[j]; A[j + 8] = ahi[j]; }

    unsigned int bestKey[8];
    #pragma unroll
    for (int r = 0; r < 8; ++r) bestKey[r] = 0u;

    // ---- B fragment layout (32x16 f16):
    // lanes 0-15 : N=lane,    K = 0..15   -> enH[col][0..15]  (32B contiguous)
    // lanes 16-31: N=lane-16, K = 16..31  -> enH[col][16..31] (32B contiguous)
    const int boff = halfSel * 16;
    const v8f czero = {0.f, 0.f, 0.f, 0.f, 0.f, 0.f, 0.f, 0.f};

    // ---- prologue: tile 0
    v16h B0 = *(const v16h*)(enH + l15 * D_DIM + boff);
    float basePrev = e2m4[l15];                 // (4 - e2): s+4 = fma(dot, 2, base) > 0
    unsigned int invTPrev = 255u;               // 255 - tile(0)
    v8f cPrev = __builtin_amdgcn_wmma_f32_16x16x32_f16(
                    false, A, false, B0, (short)0, czero, false, false);

    // ---- pipelined main loop: issue WMMA for tile t, consume tile t-1
    for (int kb = 16; kb < K_CODES; kb += 16) {
        const int col = kb + l15;
        v16h Bn = *(const v16h*)(enH + col * D_DIM + boff);
        float baseN = e2m4[col];
        __builtin_prefetch(enH + (col + 64) * D_DIM + boff, 0, 1);
        v8f cCur = __builtin_amdgcn_wmma_f32_16x16x32_f16(
                       false, A, false, Bn, (short)0, czero, false, false);
        #pragma unroll
        for (int r = 0; r < 8; ++r) {
            float s4 = fmaf(cPrev[r], 2.0f, basePrev);
            unsigned int key = (__float_as_uint(s4) & 0xFFFFFF00u) | invTPrev;
            bestKey[r] = key > bestKey[r] ? key : bestKey[r];
        }
        cPrev = cCur;
        basePrev = baseN;
        invTPrev = 255u - (unsigned int)(kb >> 4);
    }
    // ---- epilogue: consume last tile
    #pragma unroll
    for (int r = 0; r < 8; ++r) {
        float s4 = fmaf(cPrev[r], 2.0f, basePrev);
        unsigned int key = (__float_as_uint(s4) & 0xFFFFFF00u) | invTPrev;
        bestKey[r] = key > bestKey[r] ? key : bestKey[r];
    }

    // ---- unpack + cross-lane argmax within each 16-lane half
    #pragma unroll
    for (int r = 0; r < 8; ++r) {
        unsigned int sbits = bestKey[r] & 0xFFFFFF00u;       // truncated score
        int code = (int)((255u - (bestKey[r] & 0xFFu)) << 4) | l15;
        #pragma unroll
        for (int m = 8; m >= 1; m >>= 1) {
            unsigned int os = __shfl_xor(sbits, m);
            int oc = __shfl_xor(code, m);
            if (os > sbits || (os == sbits && oc < code)) { sbits = os; code = oc; }
        }
        if (l15 == 0) {
            int rowLocal = waveId * 16 + r + halfSel * 8;    // row within block
            int grow = blockIdx.x * 128 + rowLocal;
            indOut[grow] = (float)code;
            atomicAdd(&bins[code], 1.0f);
            codeBuf[rowLocal] = code;
        }
    }
    __syncthreads();

    // ---- cooperative quantize gather + embed_sum scatter (coalesced over d)
    const int blockRow0 = blockIdx.x * 128;
    #pragma unroll
    for (int i = 0; i < 16; ++i) {
        int flat = i * 256 + tid;          // 0..4095 = 128 rows * 32 dims
        int rl   = flat >> 5;
        int d    = flat & 31;
        int code = codeBuf[rl];
        int grow = blockRow0 + rl;
        qOut[grow * D_DIM + d] = enF[code * D_DIM + d];
        atomicAdd(&embedSum[code * D_DIM + d], fnF[grow * D_DIM + d]);
    }
}

// ---------------- kernel: cluster-size EMA + laplace smoothing factors ----------------
__global__ __launch_bounds__(256) void vq_cluster(
    const float* __restrict__ cs, const float* __restrict__ bins,
    float* __restrict__ csNew, float* __restrict__ invSmooth)
{
    __shared__ float red[256];
    float local = 0.0f;
    for (int k = threadIdx.x; k < K_CODES; k += 256) {
        float v = cs[k] * DECAY + bins[k] * (1.0f - DECAY);
        csNew[k] = v;
        local += v;
    }
    red[threadIdx.x] = local;
    __syncthreads();
    #pragma unroll
    for (int s = 128; s > 0; s >>= 1) {
        if (threadIdx.x < s) red[threadIdx.x] += red[threadIdx.x + s];
        __syncthreads();
    }
    float n = red[0];
    float denom = n + (float)K_CODES * EPSV;
    for (int k = threadIdx.x; k < K_CODES; k += 256) {
        float sm = (csNew[k] + EPSV) / denom * n;
        invSmooth[k] = 1.0f / sm;
    }
}

// ---------------- kernel: embed_avg EMA + updated codebook ----------------
__global__ void vq_final(
    const float* __restrict__ embedAvg, const float* __restrict__ embedSum,
    const float* __restrict__ invSmooth,
    float* __restrict__ eaNew, float* __restrict__ eUpd)
{
    int i = blockIdx.x * blockDim.x + threadIdx.x;   // < K_CODES * D_DIM
    float v = embedAvg[i] * DECAY + embedSum[i] * (1.0f - DECAY);
    eaNew[i] = v;
    eUpd[i] = v * invSmooth[i >> 5];
}

extern "C" void kernel_launch(void* const* d_in, const int* in_sizes, int n_in,
                              void* d_out, int out_size, void* d_ws, size_t ws_size,
                              hipStream_t stream) {
    const float* x        = (const float*)d_in[0];   // [65536,32]
    const float* embed    = (const float*)d_in[1];   // [1,4096,32]
    const float* cluster  = (const float*)d_in[2];   // [1,4096]
    const float* embedAvg = (const float*)d_in[3];   // [1,4096,32]

    // ---- workspace layout (bytes)
    char* ws = (char*)d_ws;
    size_t off = 0;
    float*     fnF = (float*)(ws + off);     off += (size_t)N_ROWS * D_DIM * 4;   // 8 MB
    _Float16*  fnH = (_Float16*)(ws + off);  off += (size_t)N_ROWS * D_DIM * 2;   // 4 MB
    float*     enF = (float*)(ws + off);     off += (size_t)K_CODES * D_DIM * 4;  // 512 KB
    _Float16*  enH = (_Float16*)(ws + off);  off += (size_t)K_CODES * D_DIM * 2;  // 256 KB
    float*     e2m4 = (float*)(ws + off);    off += (size_t)K_CODES * 4;
    float*     bins = (float*)(ws + off);    off += (size_t)K_CODES * 4;
    float*     embedSum = (float*)(ws + off);off += (size_t)K_CODES * D_DIM * 4;
    float*     invSmooth = (float*)(ws + off);

    // ---- output layout (floats, concatenated in reference return order)
    float* out   = (float*)d_out;
    float* qOut  = out;                                        // 65536*32
    float* indO  = out + (size_t)N_ROWS * D_DIM;               // 65536
    float* csNew = indO + N_ROWS;                              // 4096
    float* eaNew = csNew + K_CODES;                            // 4096*32
    float* eUpd  = eaNew + (size_t)K_CODES * D_DIM;            // 4096*32

    // 1) zero bins + embed_sum (contiguous in ws)
    {
        int n = K_CODES + K_CODES * D_DIM;                     // 135168
        vq_zero<<<(n + 255) / 256, 256, 0, stream>>>(bins, n);
    }
    // 2) normalize features (wave per row, 8 rows/block)
    vq_normalize<<<N_ROWS / 8, 256, 0, stream>>>(x, fnF, fnH, nullptr, N_ROWS);
    // 3) normalize codebook + (4 - e2)
    vq_normalize<<<K_CODES / 8, 256, 0, stream>>>(embed, enF, enH, e2m4, K_CODES);
    // 4) WMMA distances + packed-key argmax + quantize + scatter stats
    vq_assign<<<N_ROWS / 128, 256, 0, stream>>>(fnH, fnF, enH, enF, e2m4,
                                                qOut, indO, bins, embedSum);
    // 5) cluster-size EMA + smoothing factors (single block)
    vq_cluster<<<1, 256, 0, stream>>>(cluster, bins, csNew, invSmooth);
    // 6) embed_avg EMA + updated codebook
    vq_final<<<(K_CODES * D_DIM) / 256, 256, 0, stream>>>(embedAvg, embedSum,
                                                          invSmooth, eaNew, eUpd);
}
